// Seq2Seq_8701603742415
// MI455X (gfx1250) — compile-verified, matrix operand order checked by reading, not run on hardware
//
#include <hip/hip_runtime.h>
#include <hip/hip_bf16.h>

// ---------------------------------------------------------------------------
// Seq2Seq LSTM (2-layer encoder, 2-layer autoregressive decoder) for MI455X.
//   B=128, T_ENC=1024, IN=64, H=256, T_DEC=64, gates 4H=1024.
// Strategy: batch-split persistent kernels (8 WGs x 16 batch rows), bf16 WMMA
// (v_wmma_f32_16x16x32_bf16) with fp32 accumulation; each wave owns 16 hidden
// columns and computes its own i/f/g/o tiles so the LSTM nonlinearity runs in
// registers. Weights pre-packed to fragment-native bf16 layout (L2-resident).
// Activations use v_rcp_f32 (not IEEE divide) to shorten the serial
// recurrence critical path.
// ---------------------------------------------------------------------------

#define B_    128
#define T_ENC 1024
#define IN_   64
#define H_    256
#define T_DEC 64
#define FOURH 1024

typedef __bf16 bf16_t;
typedef __attribute__((ext_vector_type(16))) __bf16 v16bf;
typedef __attribute__((ext_vector_type(8)))  __bf16 v8bf;
typedef __attribute__((ext_vector_type(8)))  float  v8f;

// ---- workspace layout (bytes). Total = 4 MiB. -----------------------------
#define PK_E0_ELEMS (64 * 10 * 512)   /* enc L0: K=320 -> 10 k-chunks */
#define PK_512_ELEMS (64 * 16 * 512)  /* K=512 -> 16 k-chunks         */
#define OFF_E0  0
#define OFF_E1  (OFF_E0 + PK_E0_ELEMS * 2)
#define OFF_D0  (OFF_E1 + PK_512_ELEMS * 2)
#define OFF_D1  (OFF_D0 + PK_512_ELEMS * 2)
#define OFF_H0E (OFF_D1 + PK_512_ELEMS * 2)
#define OFF_H1E (OFF_H0E + B_ * H_ * 2)
#define OFF_C0E (OFF_H1E + B_ * H_ * 2)
#define OFF_C1E (OFF_C0E + B_ * H_ * 4)

// Fast activations: v_exp_f32 + v_rcp_f32, no IEEE-divide expansion.
__device__ __forceinline__ float fast_rcp(float x) {
  return __builtin_amdgcn_rcpf(x);
}
__device__ __forceinline__ float sigmoid_f(float x) {
  return fast_rcp(1.0f + __expf(-x));
}
__device__ __forceinline__ float tanh_f(float x) {
  // tanh(x) = 1 - 2/(exp(2x)+1); clamp to keep exp finite.
  x = fminf(fmaxf(x, -15.0f), 15.0f);
  float e = __expf(2.0f * x);
  return 1.0f - 2.0f * fast_rcp(e + 1.0f);
}

// A-fragment (16x32 bf16, M x K): lane L holds row M=L%16; lanes 0-15 hold
// K={0..7,16..23}, lanes 16-31 hold K={8..15,24..31}. Two 16B LDS chunks.
__device__ __forceinline__ v16bf load_a_frag(const bf16_t* act, int actS,
                                             int kbase, int lane) {
  int m  = lane & 15;
  int hi = (lane >> 4) << 3;  // 0 or 8
  const bf16_t* p = act + m * actS + kbase + hi;
  v8bf lo = *(const v8bf*)(p);
  v8bf hv = *(const v8bf*)(p + 16);
  return __builtin_shufflevector(lo, hv, 0, 1, 2, 3, 4, 5, 6, 7,
                                 8, 9, 10, 11, 12, 13, 14, 15);
}

// B-fragment: pre-packed lane-major (512 bf16 per fragment, 32B per lane).
__device__ __forceinline__ v16bf load_w_frag(const bf16_t* p) {
  v8bf lo = *(const v8bf*)(p);
  v8bf hv = *(const v8bf*)(p + 8);
  return __builtin_shufflevector(lo, hv, 0, 1, 2, 3, 4, 5, 6, 7,
                                 8, 9, 10, 11, 12, 13, 14, 15);
}

// One LSTM layer step for this block's 16-row batch tile.
// act: LDS [16][32*NKC] bf16 = [x_part ; h_part]. Each wave w computes the
// i/f/g/o 16x16 tiles for hidden cols 16w..16w+15, then updates c (LDS fp32)
// and writes new h (bf16) to up to two LDS destinations + optional fp32 out.
template <int NKC>
__device__ __forceinline__ void lstm_layer_tile(
    const bf16_t* __restrict__ act, int actS,
    const bf16_t* __restrict__ wpack,
    float* __restrict__ c_lds,
    bf16_t* __restrict__ hdst0, int hS0, int hOff0,
    bf16_t* __restrict__ hdst1, int hS1, int hOff1,
    float* __restrict__ gdst, long gRowStride,
    int lane, int w) {
  v8f a0 = {0.f, 0.f, 0.f, 0.f, 0.f, 0.f, 0.f, 0.f};
  v8f a1 = a0, a2 = a0, a3 = a0;
  const int laneOff = lane * 16;
  const bf16_t* b0p = wpack + (size_t)((0 * 16 + w) * NKC) * 512 + laneOff;
  const bf16_t* b1p = wpack + (size_t)((1 * 16 + w) * NKC) * 512 + laneOff;
  const bf16_t* b2p = wpack + (size_t)((2 * 16 + w) * NKC) * 512 + laneOff;
  const bf16_t* b3p = wpack + (size_t)((3 * 16 + w) * NKC) * 512 + laneOff;

#pragma unroll 2
  for (int kc = 0; kc < NKC; ++kc) {
    v16bf A = load_a_frag(act, actS, kc * 32, lane);
    v16bf B0 = load_w_frag(b0p + (size_t)kc * 512);
    a0 = __builtin_amdgcn_wmma_f32_16x16x32_bf16(false, A, false, B0,
                                                 (short)0, a0, false, false);
    v16bf B1 = load_w_frag(b1p + (size_t)kc * 512);
    a1 = __builtin_amdgcn_wmma_f32_16x16x32_bf16(false, A, false, B1,
                                                 (short)0, a1, false, false);
    v16bf B2 = load_w_frag(b2p + (size_t)kc * 512);
    a2 = __builtin_amdgcn_wmma_f32_16x16x32_bf16(false, A, false, B2,
                                                 (short)0, a2, false, false);
    v16bf B3 = load_w_frag(b3p + (size_t)kc * 512);
    a3 = __builtin_amdgcn_wmma_f32_16x16x32_bf16(false, A, false, B3,
                                                 (short)0, a3, false, false);
  }

  // All waves must finish reading `act` before anyone overwrites the h region.
  __syncthreads();

  // C/D layout: lane L, VGPR r -> M = r + (L>=16 ? 8 : 0), N = L%16.
  int j  = (w << 4) + (lane & 15);
  int mb = (lane >> 4) << 3;
#pragma unroll
  for (int r = 0; r < 8; ++r) {
    int m = mb + r;
    float cold = c_lds[m * H_ + j];
    float iv = sigmoid_f(a0[r]);
    float fv = sigmoid_f(a1[r]);
    float gv = tanh_f(a2[r]);
    float ov = sigmoid_f(a3[r]);
    float cn = fv * cold + iv * gv;
    float hn = ov * tanh_f(cn);
    c_lds[m * H_ + j] = cn;
    hdst0[m * hS0 + hOff0 + j] = (bf16_t)hn;
    if (hdst1) hdst1[m * hS1 + hOff1 + j] = (bf16_t)hn;
    if (gdst) gdst[(long)m * gRowStride + j] = hn;
  }
}

// ---------------------------------------------------------------------------
// Prep: fp32 weights -> packed bf16 B-fragments. W = concat([Wa(Ka) Wb(Kb)])
// along K; fragment(n, kc) is 512 bf16 lane-major matching the 16-bit
// 32x16 B VGPR striping (lanes 0-15: K={0..7,16..23}; lanes 16-31: +8).
// ---------------------------------------------------------------------------
__global__ void pack_weights_kernel(const float* __restrict__ Wa, int Ka,
                                    const float* __restrict__ Wb, int Kb,
                                    bf16_t* __restrict__ dst, int nElems,
                                    int nkc) {
  int idx = blockIdx.x * blockDim.x + threadIdx.x;
  if (idx >= nElems) return;
  int e = idx & 511;
  int lane = e >> 4;
  int j = e & 15;
  int fragIdx = idx >> 9;
  int kc = fragIdx % nkc;
  int n = fragIdx / nkc;
  int koff = (j < 8 ? j : j + 8) + ((lane >= 16) ? 8 : 0);
  int k = kc * 32 + koff;
  int row = n * 16 + (lane & 15);
  float v = (k < Ka) ? Wa[row * Ka + k] : Wb[row * Kb + (k - Ka)];
  dst[idx] = (bf16_t)v;
}

// ---------------------------------------------------------------------------
// Encoder: persistent, 8 blocks x 512 threads (16 waves). Block bb owns batch
// rows 16bb..16bb+15. LDS: xh0 [16][320] bf16 (x_t | h0), xh1 [16][512] bf16
// (h0_new | h1), c0/c1 [16][256] fp32.
// ---------------------------------------------------------------------------
__global__ void __launch_bounds__(512, 1) lstm_encoder_kernel(
    const float* __restrict__ x, const bf16_t* __restrict__ wp_e0,
    const bf16_t* __restrict__ wp_e1, float* __restrict__ enc_out,
    bf16_t* __restrict__ h0e, bf16_t* __restrict__ h1e,
    float* __restrict__ c0e, float* __restrict__ c1e) {
  extern __shared__ char smem[];
  bf16_t* xh0 = (bf16_t*)smem;                         // 16*320*2 = 10240 B
  bf16_t* xh1 = (bf16_t*)(smem + 10240);               // 16*512*2 = 16384 B
  float* c0 = (float*)(smem + 10240 + 16384);          // 16*256*4 = 16384 B
  float* c1 = (float*)(smem + 10240 + 16384 + 16384);  // 16384 B

  const int tid = threadIdx.x;
  const int lane = tid & 31;
  const int w = tid >> 5;
  const int bb = blockIdx.x;

  for (int i = tid; i < 16 * H_; i += 512) { c0[i] = 0.f; c1[i] = 0.f; }
  for (int i = tid; i < 16 * 320; i += 512) xh0[i] = (bf16_t)0.f;
  for (int i = tid; i < 16 * 512; i += 512) xh1[i] = (bf16_t)0.f;
  __syncthreads();

  for (int t = 0; t < T_ENC; ++t) {
    // stage x_t (bf16) into xh0[:, 0:64]; prefetch next timestep's rows
    for (int i = tid; i < 16 * IN_; i += 512) {
      int m = i / IN_, k = i % IN_;
      const float* src = &x[((size_t)(bb * 16 + m) * T_ENC + t) * IN_ + k];
      xh0[m * 320 + k] = (bf16_t)(*src);
      if (t + 1 < T_ENC) __builtin_prefetch(src + IN_, 0, 0);
    }
    __syncthreads();

    // layer 0: [x_t ; h0] (K=320) -> h0_new into xh0 h-slot + xh1 input slot
    lstm_layer_tile<10>(xh0, 320, wp_e0, c0,
                        xh0, 320, 64, xh1, 512, 0,
                        (float*)nullptr, 0, lane, w);
    __syncthreads();

    // layer 1: [h0_new ; h1] (K=512) -> h1_new into xh1 h-slot + enc_out
    lstm_layer_tile<16>(xh1, 512, wp_e1, c1,
                        xh1, 512, 256, (bf16_t*)nullptr, 0, 0,
                        enc_out + ((size_t)(bb * 16) * T_ENC + t) * H_,
                        (long)T_ENC * H_, lane, w);
    __syncthreads();
  }

  // hand final (h, c) of both layers to the decoder kernel
  for (int i = tid; i < 16 * H_; i += 512) {
    int m = i / H_, j = i % H_;
    int g = (bb * 16 + m) * H_ + j;
    h0e[g] = xh0[m * 320 + 64 + j];
    h1e[g] = xh1[m * 512 + 256 + j];
    c0e[g] = c0[i];
    c1e[g] = c1[i];
  }
}

// ---------------------------------------------------------------------------
// Decoder: autoregressive, y_{t+1} = h1_new. LDS: xh0 [16][512] (y | h0),
// xh1 [16][512] (h0_new | h1), c0/c1 fp32.
// ---------------------------------------------------------------------------
__global__ void __launch_bounds__(512, 1) lstm_decoder_kernel(
    const bf16_t* __restrict__ wp_d0, const bf16_t* __restrict__ wp_d1,
    float* __restrict__ dec_out, const bf16_t* __restrict__ h0e,
    const bf16_t* __restrict__ h1e, const float* __restrict__ c0e,
    const float* __restrict__ c1e) {
  extern __shared__ char smem[];
  bf16_t* xh0 = (bf16_t*)smem;                 // 16384 B
  bf16_t* xh1 = (bf16_t*)(smem + 16384);       // 16384 B
  float* c0 = (float*)(smem + 32768);          // 16384 B
  float* c1 = (float*)(smem + 32768 + 16384);  // 16384 B

  const int tid = threadIdx.x;
  const int lane = tid & 31;
  const int w = tid >> 5;
  const int bb = blockIdx.x;

  // init: y = h1e (last encoder output), states from encoder finals
  for (int i = tid; i < 16 * H_; i += 512) {
    int m = i / H_, j = i % H_;
    int g = (bb * 16 + m) * H_ + j;
    xh0[m * 512 + j] = h1e[g];
    xh0[m * 512 + 256 + j] = h0e[g];
    xh1[m * 512 + j] = (bf16_t)0.f;
    xh1[m * 512 + 256 + j] = h1e[g];
    c0[i] = c0e[g];
    c1[i] = c1e[g];
  }
  __syncthreads();

  for (int t = 0; t < T_DEC; ++t) {
    // layer 0: [y ; h0] (K=512) -> h0_new into xh0 h-slot + xh1 input slot
    lstm_layer_tile<16>(xh0, 512, wp_d0, c0,
                        xh0, 512, 256, xh1, 512, 0,
                        (float*)nullptr, 0, lane, w);
    __syncthreads();

    // layer 1: [h0_new ; h1] -> h1_new into xh1 h-slot, next y, dec_out
    lstm_layer_tile<16>(xh1, 512, wp_d1, c1,
                        xh1, 512, 256, xh0, 512, 0,
                        dec_out + ((size_t)(bb * 16) * T_DEC + t) * H_,
                        (long)T_DEC * H_, lane, w);
    __syncthreads();
  }
}

// ---------------------------------------------------------------------------
extern "C" void kernel_launch(void* const* d_in, const int* in_sizes, int n_in,
                              void* d_out, int out_size, void* d_ws,
                              size_t ws_size, hipStream_t stream) {
  (void)in_sizes; (void)n_in; (void)out_size; (void)ws_size;
  const float* x = (const float*)d_in[0];
  const float* eWih0 = (const float*)d_in[1];
  const float* eWhh0 = (const float*)d_in[2];
  const float* eWih1 = (const float*)d_in[3];
  const float* eWhh1 = (const float*)d_in[4];
  const float* dWih0 = (const float*)d_in[5];
  const float* dWhh0 = (const float*)d_in[6];
  const float* dWih1 = (const float*)d_in[7];
  const float* dWhh1 = (const float*)d_in[8];

  char* ws = (char*)d_ws;
  bf16_t* wp_e0 = (bf16_t*)(ws + OFF_E0);
  bf16_t* wp_e1 = (bf16_t*)(ws + OFF_E1);
  bf16_t* wp_d0 = (bf16_t*)(ws + OFF_D0);
  bf16_t* wp_d1 = (bf16_t*)(ws + OFF_D1);
  bf16_t* h0e = (bf16_t*)(ws + OFF_H0E);
  bf16_t* h1e = (bf16_t*)(ws + OFF_H1E);
  float* c0e = (float*)(ws + OFF_C0E);
  float* c1e = (float*)(ws + OFF_C1E);

  // pack weights to fragment-native bf16 (once per launch; cheap)
  pack_weights_kernel<<<(PK_E0_ELEMS + 255) / 256, 256, 0, stream>>>(
      eWih0, IN_, eWhh0, H_, wp_e0, PK_E0_ELEMS, 10);
  pack_weights_kernel<<<(PK_512_ELEMS + 255) / 256, 256, 0, stream>>>(
      eWih1, H_, eWhh1, H_, wp_e1, PK_512_ELEMS, 16);
  pack_weights_kernel<<<(PK_512_ELEMS + 255) / 256, 256, 0, stream>>>(
      dWih0, H_, dWhh0, H_, wp_d0, PK_512_ELEMS, 16);
  pack_weights_kernel<<<(PK_512_ELEMS + 255) / 256, 256, 0, stream>>>(
      dWih1, H_, dWhh1, H_, wp_d1, PK_512_ELEMS, 16);

  float* enc_out = (float*)d_out;                            // [B,T_ENC,H]
  float* dec_out = enc_out + (size_t)B_ * T_ENC * H_;        // [B,T_DEC,H]

  size_t smemE = 10240 + 16384 + 16384 + 16384;  // 59392 B
  lstm_encoder_kernel<<<B_ / 16, 512, smemE, stream>>>(
      x, wp_e0, wp_e1, enc_out, h0e, h1e, c0e, c1e);

  size_t smemD = 16384 * 4;  // 65536 B (well under 320 KB/WGP)
  lstm_decoder_kernel<<<B_ / 16, 512, smemD, stream>>>(
      wp_d0, wp_d1, dec_out, h0e, h1e, c0e, c1e);
}
